// ConvSelfAttention_26104811225680
// MI455X (gfx1250) — compile-verified, matrix operand order checked by reading, not run on hardware
//
#include <hip/hip_runtime.h>

// ---------------------------------------------------------------------------
// ConvSelfAttention for MI455X (gfx1250, wave32, WMMA + TDM).
//   inputs [8,512,64,64] f32, Wa [384,512], Wo [512,256], gamma [1]
//   All GEMMs via v_wmma_f32_16x16x32_f16 (fp16 operands, fp32 accum).
//   Flash-style streaming softmax (never materializes the 512 MB S matrix).
//   V tiles are fed to the attention loop by the Tensor Data Mover
//   (tensor_load_to_lds, double-buffered, s_wait_tensorcnt) so the DMA
//   overlaps the S-tile WMMAs and softmax VALU work.
//   Q/K/V/Yt intermediates (40 MB fp16) live in d_ws, resident in 192 MB L2.
// ---------------------------------------------------------------------------

typedef _Float16 half_t;
typedef __attribute__((ext_vector_type(16))) _Float16      v16h;
typedef __attribute__((ext_vector_type(8)))  float         v8f;
typedef __attribute__((ext_vector_type(4)))  unsigned int  v4u;
typedef __attribute__((ext_vector_type(8)))  int           v8i_t;
typedef __attribute__((ext_vector_type(4)))  int           v4i_t;

#define BATCH 8
#define C_IN  512     // input channels
#define HC    64      // query/key channels
#define CV    256     // value channels (C_IN/2)
#define NPIX  4096    // 64*64 spatial positions

static __device__ __forceinline__ v8f wmma_f16(v16h a, v16h b, v8f c) {
  // D = A(16x32 f16) * B(32x16 f16) + C(16x16 f32)
  return __builtin_amdgcn_wmma_f32_16x16x32_f16(
      /*neg_a=*/false, a, /*neg_b=*/false, b,
      /*c_mod=*/(short)0, c, /*reuse_a=*/false, /*reuse_b=*/false);
}

static __device__ __forceinline__ float rowmax16(float x) {
  x = fmaxf(x, __shfl_xor(x, 1, 32));
  x = fmaxf(x, __shfl_xor(x, 2, 32));
  x = fmaxf(x, __shfl_xor(x, 4, 32));
  x = fmaxf(x, __shfl_xor(x, 8, 32));
  return x;
}
static __device__ __forceinline__ float rowsum16(float x) {
  x += __shfl_xor(x, 1, 32);
  x += __shfl_xor(x, 2, 32);
  x += __shfl_xor(x, 4, 32);
  x += __shfl_xor(x, 8, 32);
  return x;
}

// ---------------------------------------------------------------------------
// TDM: DMA a 256(row) x 32(col) fp16 tile of V (row stride 4096 elements in
// global memory) into LDS, packed dense as [256][32]. D# layout per
// CDNA5 ISA (2D tile, groups 2/3 zero).
// ---------------------------------------------------------------------------
static __device__ __forceinline__ void tdm_load_v_tile(const half_t* gsrc,
                                                       unsigned lds_off) {
  unsigned long long ga = (unsigned long long)(size_t)gsrc;
  v4u g0;
  g0[0] = 1u;                                           // count=1, user mode
  g0[1] = lds_off;                                      // lds_addr (bytes)
  g0[2] = (unsigned)(ga & 0xffffffffu);                 // global_addr[31:0]
  g0[3] = (unsigned)((ga >> 32) & 0x01ffffffu)          // global_addr[56:32]
          | (2u << 30);                                 // type = 2 ("image")
  v8i_t g1;
  g1[0] = (int)(1u << 16);        // workgroup_mask=0, data_size=1 (2 bytes)
  g1[1] = (int)(4096u << 16);     // tensor_dim0 = 4096 (bits 79:48, low half)
  g1[2] = (int)(256u << 16);      // tensor_dim0 hi=0 | tensor_dim1 = 256
  g1[3] = (int)(32u << 16);       // tensor_dim1 hi=0 | tile_dim0 = 32
  g1[4] = 256;                    // tile_dim1 = 256, tile_dim2 = 0
  g1[5] = 4096;                   // tensor_dim0_stride = 4096 (low 32)
  g1[6] = 0;                      // stride hi | tensor_dim1_stride lo (2D)
  g1[7] = 0;
  v4i_t gz = {0, 0, 0, 0};
#if __clang_major__ >= 23
  v8i_t gz8 = {0, 0, 0, 0, 0, 0, 0, 0};
  __builtin_amdgcn_tensor_load_to_lds(g0, g1, gz, gz, gz8, 0);
#else
  __builtin_amdgcn_tensor_load_to_lds(g0, g1, gz, gz, 0);
#endif
}

// ---------------------------------------------------------------------------
// Kernel 1: QKV projection  W = Wa * X   (per batch, X = inputs[b] [512,4096])
//   Writes Qt [B,4096,64] f16, Kt [B,4096,64] f16 (transposed so attention
//   A/B tiles are contiguous), V [B,256,4096] f16 (natural).
//   One wave computes a 16(o) x 64(n) tile. Each lane owns one k-row of the
//   X tile and loads it as 4-wide b128, transposing into LDS as f16.
//   The Q/K/V scatter branches on the SCALAR oBase (a 16-row o-tile never
//   straddles the 64/128 region boundaries) -> pure SALU branch, no exec
//   mask juggling around the stores.
// ---------------------------------------------------------------------------
__global__ void __launch_bounds__(32)
qkv_proj_kernel(const float* __restrict__ x, const float* __restrict__ Wa,
                half_t* __restrict__ Qt, half_t* __restrict__ Kt,
                half_t* __restrict__ V) {
  __shared__ half_t Xs[64 * 32];          // [n 64][k 32] f16, 4 KB
  const int lane  = threadIdx.x;
  const int lm    = lane & 15;
  const int lh    = lane >> 4;            // 0/1 half-group
  const int nBase = blockIdx.x * 64;
  const int oBase = blockIdx.y * 16;
  const int b     = blockIdx.z;
  const float* xb = x + (size_t)b * C_IN * NPIX;

  v8f acc[4] = {};
  for (int k0 = 0; k0 < C_IN; k0 += 32) {
    __syncthreads();
    // lane owns global row k0+lane: 64 floats as 16x float4, transpose to LDS
    const float* xrow = xb + (size_t)(k0 + lane) * NPIX + nBase;
#pragma unroll
    for (int j4 = 0; j4 < 16; ++j4) {
      float4 f = *(const float4*)(xrow + j4 * 4);
      Xs[(j4 * 4 + 0) * 32 + lane] = (half_t)f.x;
      Xs[(j4 * 4 + 1) * 32 + lane] = (half_t)f.y;
      Xs[(j4 * 4 + 2) * 32 + lane] = (half_t)f.z;
      Xs[(j4 * 4 + 3) * 32 + lane] = (half_t)f.w;
    }
    __syncthreads();
    // A tile: Wa[oBase+lm, k0 + lh*16 + j]  (contiguous f32 -> f16)
    v16h a;
    const float* wrow = Wa + (size_t)(oBase + lm) * C_IN + k0 + lh * 16;
#pragma unroll
    for (int j = 0; j < 16; ++j) a[j] = (half_t)wrow[j];
#pragma unroll
    for (int jt = 0; jt < 4; ++jt) {
      v16h bb = *(const v16h*)(&Xs[(jt * 16 + lm) * 32 + lh * 16]);
      acc[jt] = wmma_f16(a, bb, acc[jt]);
    }
  }

  // scatter with UNIFORM region selection (oBase is scalar)
  const int oLoc = lh * 8;                // per-lane row offset base (+v)
  if (oBase < HC) {
    half_t* dst = Qt + ((size_t)b * NPIX + nBase + lm) * HC + oBase;
#pragma unroll
    for (int jt = 0; jt < 4; ++jt)
#pragma unroll
      for (int v = 0; v < 8; ++v)
        dst[(size_t)(jt * 16) * HC + oLoc + v] = (half_t)acc[jt][v];
  } else if (oBase < 2 * HC) {
    half_t* dst = Kt + ((size_t)b * NPIX + nBase + lm) * HC + (oBase - HC);
#pragma unroll
    for (int jt = 0; jt < 4; ++jt)
#pragma unroll
      for (int v = 0; v < 8; ++v)
        dst[(size_t)(jt * 16) * HC + oLoc + v] = (half_t)acc[jt][v];
  } else {
    half_t* dst = V + ((size_t)b * CV + (oBase - 2 * HC)) * NPIX + nBase + lm;
#pragma unroll
    for (int jt = 0; jt < 4; ++jt)
#pragma unroll
      for (int v = 0; v < 8; ++v)
        dst[(size_t)(oLoc + v) * NPIX + jt * 16] = (half_t)acc[jt][v];
  }
}

// ---------------------------------------------------------------------------
// Kernel 2: flash attention. One wave owns a 16-row n-block:
//   stream m in chunks of 32: S tile = Q^T K (2x2 wmma over d=64),
//   running max/sum softmax, P bounced via 1 KB LDS into A-layout.
//   V chunks (256x32 f16 tiles) arrive via TDM into double-buffered LDS,
//   overlapped with compute and synchronized with s_wait_tensorcnt.
//   Y accumulators: 16 c-tiles (all 256 value channels) of 16x16 f32.
//   Writes Yt [B,4096,256] f16 (transposed, lane-coalesced).
// ---------------------------------------------------------------------------
__global__ void __launch_bounds__(32)
attn_kernel(const half_t* __restrict__ Qt, const half_t* __restrict__ Kt,
            const half_t* __restrict__ V, half_t* __restrict__ Yt) {
  __shared__ half_t Vs[2 * CV * 32];      // double-buffered TDM dest, 32 KB
  __shared__ half_t Pst[16 * 32];         // softmax tile staging (f16), 1 KB
  const int lane  = threadIdx.x;
  const int lm    = lane & 15;
  const int lh    = lane >> 4;
  const int nBase = blockIdx.x * 16;
  const int b     = blockIdx.y;
  const half_t* qb = Qt + (size_t)b * NPIX * HC;
  const half_t* kb = Kt + (size_t)b * NPIX * HC;
  const half_t* vb = V  + (size_t)b * CV * NPIX;
  const unsigned vs_off0 = (unsigned)(size_t)(&Vs[0]);
  const unsigned vs_off1 = (unsigned)(size_t)(&Vs[CV * 32]);

  // resident Q^T A-tiles: rows nBase+lm, c chunks 0-31 / 32-63
  v16h aq0 = *(const v16h*)(qb + (size_t)(nBase + lm) * HC + lh * 16);
  v16h aq1 = *(const v16h*)(qb + (size_t)(nBase + lm) * HC + lh * 16 + 32);

  v8f acc[16] = {};                       // Y accumulators: 256 c in 16 tiles
  float mrun[8], lrun[8];
#pragma unroll
  for (int v = 0; v < 8; ++v) { mrun[v] = -3.0e38f; lrun[v] = 0.0f; }

  // prime the pipeline: chunk 0 -> buffer 0
  tdm_load_v_tile(vb, vs_off0);

  for (int m0 = 0; m0 < NPIX; m0 += 32) {
    const int buf = (m0 >> 5) & 1;
    const half_t* krow = kb + (size_t)m0 * HC;
    if (m0 + 32 < NPIX) {
      // pull next K chunk toward the caches (global_prefetch_b8)
      __builtin_prefetch(krow + 32 * HC + lane * 64, 0, 0);
    }
    // K B-tiles: B[K=c, N=m], lane holds column m = m0(+16)+lm, contiguous c
    v16h bk00 = *(const v16h*)(krow + (size_t)lm * HC + lh * 16);
    v16h bk01 = *(const v16h*)(krow + (size_t)lm * HC + lh * 16 + 32);
    v16h bk10 = *(const v16h*)(krow + (size_t)(16 + lm) * HC + lh * 16);
    v16h bk11 = *(const v16h*)(krow + (size_t)(16 + lm) * HC + lh * 16 + 32);

    v8f s0 = {}, s1 = {};
    s0 = wmma_f16(aq0, bk00, s0); s0 = wmma_f16(aq1, bk01, s0);
    s1 = wmma_f16(aq0, bk10, s1); s1 = wmma_f16(aq1, bk11, s1);

    // streaming softmax update (row r = v + 8*lh, cols across 16 lanes)
    float p0[8], p1[8];
#pragma unroll
    for (int v = 0; v < 8; ++v) {
      float cm   = rowmax16(fmaxf(s0[v], s1[v]));
      float mnew = fmaxf(mrun[v], cm);
      float scl  = __expf(mrun[v] - mnew);
      mrun[v] = mnew;
      p0[v] = __expf(s0[v] - mnew);
      p1[v] = __expf(s1[v] - mnew);
      lrun[v] = lrun[v] * scl + rowsum16(p0[v] + p1[v]);
#pragma unroll
      for (int ct = 0; ct < 16; ++ct) acc[ct][v] *= scl;
    }

    // re-layout P (C/D form) -> A form via LDS
    __syncthreads();
#pragma unroll
    for (int v = 0; v < 8; ++v) {
      int n = v + 8 * lh;
      Pst[n * 32 + lm]      = (half_t)p0[v];
      Pst[n * 32 + 16 + lm] = (half_t)p1[v];
    }
    __syncthreads();
    v16h ap = *(const v16h*)(&Pst[lm * 32 + lh * 16]);

    // kick off next V chunk into the other buffer, then wait for current one
    if (m0 + 32 < NPIX) {
      tdm_load_v_tile(vb + m0 + 32, buf ? vs_off0 : vs_off1);
      __builtin_amdgcn_s_wait_tensorcnt(1);   // current buffer complete
    } else {
      __builtin_amdgcn_s_wait_tensorcnt(0);   // drain
    }

    // Y += P(16x32) * V-tiles(32x16) for all 16 c-tiles, V from LDS [c][32]
    const half_t* vtile = &Vs[buf * CV * 32];
#pragma unroll
    for (int ct = 0; ct < 16; ++ct) {
      v16h bv = *(const v16h*)(vtile + (ct * 16 + lm) * 32 + lh * 16);
      acc[ct] = wmma_f16(ap, bv, acc[ct]);
    }
  }

  // finalize: one reciprocal per row, then multiply; store Yt[n, c] coalesced
  float rinv[8];
#pragma unroll
  for (int v = 0; v < 8; ++v) rinv[v] = 1.0f / lrun[v];
#pragma unroll
  for (int ct = 0; ct < 16; ++ct) {
#pragma unroll
    for (int v = 0; v < 8; ++v) {
      int n = nBase + v + 8 * lh;
      int c = ct * 16 + lm;
      Yt[((size_t)b * NPIX + n) * CV + c] = (half_t)(acc[ct][v] * rinv[v]);
    }
  }
}

// ---------------------------------------------------------------------------
// Kernel 3: O = gamma * (Wo * Y) + X. Both A (Wo rows) and B (Yt rows) are
// already contiguous in K, no LDS needed. Fused residual, coalesced f32 store.
// ---------------------------------------------------------------------------
__global__ void __launch_bounds__(32)
out_proj_kernel(const half_t* __restrict__ Yt, const float* __restrict__ Wo,
                const float* __restrict__ x, const float* __restrict__ gamma,
                float* __restrict__ out) {
  const int lane  = threadIdx.x;
  const int lm    = lane & 15;
  const int lh    = lane >> 4;
  const int nBase = blockIdx.x * 64;
  const int oBase = blockIdx.y * 16;
  const int b     = blockIdx.z;
  const half_t* yb = Yt + (size_t)b * NPIX * CV;

  v8f acc[4] = {};
  for (int k0 = 0; k0 < CV; k0 += 32) {
    v16h a;
    const float* wrow = Wo + (size_t)(oBase + lm) * CV + k0 + lh * 16;
#pragma unroll
    for (int j = 0; j < 16; ++j) a[j] = (half_t)wrow[j];
#pragma unroll
    for (int jt = 0; jt < 4; ++jt) {
      v16h bb = *(const v16h*)(yb + (size_t)(nBase + jt * 16 + lm) * CV + k0 + lh * 16);
      acc[jt] = wmma_f16(a, bb, acc[jt]);
    }
  }
  const float g = gamma[0];
#pragma unroll
  for (int jt = 0; jt < 4; ++jt) {
#pragma unroll
    for (int v = 0; v < 8; ++v) {
      int o = oBase + v + 8 * lh;
      int n = nBase + jt * 16 + lm;
      size_t idx = ((size_t)b * C_IN + o) * NPIX + n;
      out[idx] = g * acc[jt][v] + x[idx];     // fused residual, coalesced
    }
  }
}

// ---------------------------------------------------------------------------
extern "C" void kernel_launch(void* const* d_in, const int* in_sizes, int n_in,
                              void* d_out, int out_size, void* d_ws, size_t ws_size,
                              hipStream_t stream) {
  (void)in_sizes; (void)n_in; (void)out_size; (void)ws_size;
  const float* x     = (const float*)d_in[0];   // [8,512,64,64]
  const float* Wa    = (const float*)d_in[1];   // [384,512]
  const float* Wo    = (const float*)d_in[2];   // [512,256]
  const float* gamma = (const float*)d_in[3];   // [1]
  float* out = (float*)d_out;

  // workspace layout (fp16, 40 MB total; fully overwritten before any read)
  char* ws = (char*)d_ws;
  half_t* Qt = (half_t*)(ws);                        //  4 MB: [8,4096,64]
  half_t* Kt = (half_t*)(ws + ((size_t)4  << 20));   //  4 MB: [8,4096,64]
  half_t* V  = (half_t*)(ws + ((size_t)8  << 20));   // 16 MB: [8,256,4096]
  half_t* Yt = (half_t*)(ws + ((size_t)24 << 20));   // 16 MB: [8,4096,256]

  qkv_proj_kernel<<<dim3(64, 24, 8), 32, 0, stream>>>(x, Wa, Qt, Kt, V);
  attn_kernel    <<<dim3(256, 8),     32, 0, stream>>>(Qt, Kt, V, Yt);
  out_proj_kernel<<<dim3(64, 32, 8),  32, 0, stream>>>(Yt, Wo, x, gamma, out);
}
